// MLAttention_28845000360007
// MI455X (gfx1250) — matrix operand order repaired and checked by reading, not compiled, not run on hardware
//
#include <hip/hip_runtime.h>
#include <hip/hip_bf16.h>

typedef __bf16 bf16;
typedef __attribute__((ext_vector_type(16))) __bf16 v16bf;
typedef __attribute__((ext_vector_type(8)))  float  v8f;
typedef __attribute__((ext_vector_type(4)))  unsigned int v4u;
typedef __attribute__((ext_vector_type(8)))  int v8i;
typedef __attribute__((ext_vector_type(4)))  int v4i;

#define S_LEN 2048
#define HID   2048
#define NH    16
#define HDIM  128
#define DRT   64
#define DQK   192
#define CKV_D 512
#define CQ_D  1536

#if __has_builtin(__builtin_amdgcn_tensor_load_to_lds) && \
    __has_builtin(__builtin_amdgcn_s_wait_tensorcnt)
#define HAVE_TDM 1
#endif

// All fragments are built from two 16-byte loads (LDS: ds_load_b128).
union Frag { v16bf v; uint4 q[2]; };
__device__ inline v16bf load_frag_pair(const bf16* p0, const bf16* p1) {
  Frag f;
  f.q[0] = *(const uint4*)p0;
  f.q[1] = *(const uint4*)p1;
  return f.v;
}

#ifdef HAVE_TDM
// Issue a 2D TDM load: tile (tile_d0 x tile_d1) of 2-byte elements from a
// row-major tensor (row length tensor_d0, row count tensor_d1 from the tile
// start, row stride stride_d0 elements) into contiguous LDS at lds_off.
// D# layout per CDNA5 ISA ch.8 (group0: 4 dw, group1: 8 dw).
// 6-arg builtin form (clang-23 / therock-10.0).
__device__ inline void tdm_load_2d(unsigned lds_off, const void* gaddr,
                                   unsigned tile_d0, unsigned tile_d1,
                                   unsigned tensor_d0, unsigned tensor_d1,
                                   unsigned long stride_d0) {
  unsigned long ga = (unsigned long)gaddr;
  v4u g0;
  g0[0] = 1u;                                        // count=1 (valid), user mode
  g0[1] = lds_off;                                   // lds_addr (bytes)
  g0[2] = (unsigned)(ga & 0xffffffffu);              // global_addr[31:0]
  g0[3] = (unsigned)((ga >> 32) & 0x01ffffffu)       // global_addr[56:32]
        | (2u << 30);                                // type = 2 ("image")
  v8i g1;
  g1[0] = (int)(1u << 16);                           // data_size = 1 (2 bytes)
  g1[1] = (int)((tensor_d0 & 0xffffu) << 16);        // tensor_dim0[15:0]
  g1[2] = (int)(((tensor_d0 >> 16) & 0xffffu) |
                ((tensor_d1 & 0xffffu) << 16));      // td0[31:16] | td1[15:0]
  g1[3] = (int)(((tensor_d1 >> 16) & 0xffffu) |
                ((tile_d0 & 0xffffu) << 16));        // td1[31:16] | tile_dim0
  g1[4] = (int)(tile_d1 & 0xffffu);                  // tile_dim1 (tile_dim2 = 0)
  g1[5] = (int)(stride_d0 & 0xffffffffu);            // tensor_dim0_stride[31:0]
  g1[6] = (int)((stride_d0 >> 32) & 0xffffu);        // stride[47:32]
  g1[7] = 0;
  v4i zz4 = {0, 0, 0, 0};
  v8i zz8 = {0, 0, 0, 0, 0, 0, 0, 0};
  __builtin_amdgcn_tensor_load_to_lds(g0, g1, zz4, zz4, zz8, 0);
}
#endif

// ---------------------------------------------------------------- casts
__global__ void cast_f32_bf16(const float* __restrict__ s, bf16* __restrict__ d, int n) {
  int i = blockIdx.x * blockDim.x + threadIdx.x;
  if (i < n) d[i] = (bf16)s[i];
}
// s: [R][C] fp32 row-major -> d: [C][R] bf16 (K-major weight layout for WMMA B).
__global__ void cast_transpose_f32_bf16(const float* __restrict__ s,
                                        bf16* __restrict__ d, int R, int C) {
  int idx = blockIdx.x * blockDim.x + threadIdx.x;
  if (idx < R * C) {
    int r = idx / C, c = idx % C;
    d[(size_t)c * R + r] = (bf16)s[idx];
  }
}

// ---------------------------------------------------------------- tiled bf16 GEMM
// C[M,N] = A[M,K] x Bt[N,K]^T, fp32 accumulate. 256 threads = 8 waves,
// tile 128x128, K-step 32, wave grid 4(M) x 2(N), 8 WMMA tiles per wave.
// TDM double-buffered: tile t+1 DMA overlaps tile t compute
// (2 loads per tile in flight; TDM retires in-order -> wait tensorcnt<=2).
#define BM 128
#define BN 128
#define BK 32

__global__ __launch_bounds__(256) void gemm_bf16_kernel(
    const bf16* __restrict__ A, const bf16* __restrict__ Bt,
    void* __restrict__ Cout, int M, int N, int K, int c_is_bf16)
{
  __shared__ __align__(16) bf16 sA[2][BM][BK];    // 2 x 8 KB ping-pong
  __shared__ __align__(16) bf16 sBt[2][BN][BK];   // 2 x 8 KB ping-pong

  const int tid  = threadIdx.x;
  const int w    = tid >> 5;
  const int l    = tid & 31;
  const int wm   = w >> 1;
  const int wn   = w & 1;
  const int lm   = l & 15;
  const int half = l >> 4;
  const int bm0  = blockIdx.y * BM;
  const int bn0  = blockIdx.x * BN;
  const int niter = K / BK;

  const v8f zero8 = {0.f,0.f,0.f,0.f,0.f,0.f,0.f,0.f};
  v8f acc[2][4];
#pragma unroll
  for (int mt = 0; mt < 2; ++mt)
#pragma unroll
    for (int nt = 0; nt < 4; ++nt) acc[mt][nt] = zero8;

#ifdef HAVE_TDM
  if (w == 0) {   // prologue: DMA tile 0
    tdm_load_2d((unsigned)(size_t)&sA[0][0][0],  A  + (size_t)bm0 * K,
                BK, BM, (unsigned)K, (unsigned)(M - bm0), (unsigned long)K);
    tdm_load_2d((unsigned)(size_t)&sBt[0][0][0], Bt + (size_t)bn0 * K,
                BK, BN, (unsigned)K, (unsigned)(N - bn0), (unsigned long)K);
  }
#endif

  for (int it = 0; it < niter; ++it) {
    const int cur = it & 1;
    const int k0  = it * BK;
#ifdef HAVE_TDM
    if (w == 0) {
      if (it + 1 < niter) {   // DMA next tile into the other buffer, then
        const int nk = k0 + BK;   // wait only for the *current* tile's 2 loads
        tdm_load_2d((unsigned)(size_t)&sA[1 - cur][0][0],  A  + (size_t)bm0 * K + nk,
                    BK, BM, (unsigned)(K - nk), (unsigned)(M - bm0), (unsigned long)K);
        tdm_load_2d((unsigned)(size_t)&sBt[1 - cur][0][0], Bt + (size_t)bn0 * K + nk,
                    BK, BN, (unsigned)(K - nk), (unsigned)(N - bn0), (unsigned long)K);
        __builtin_amdgcn_s_wait_tensorcnt(2);
      } else {
        __builtin_amdgcn_s_wait_tensorcnt(0);
      }
    }
#else
    for (int ch = tid; ch < (BM * BK) / 8; ch += 256) {
      int row = ch >> 2, c8 = (ch & 3) * 8;
      *(uint4*)&sA[cur][row][c8] = *(const uint4*)(A + (size_t)(bm0 + row) * K + k0 + c8);
    }
    for (int ch = tid; ch < (BN * BK) / 8; ch += 256) {
      int row = ch >> 2, c8 = (ch & 3) * 8;
      uint4 val = {0u, 0u, 0u, 0u};
      if (bn0 + row < N)
        val = *(const uint4*)(Bt + (size_t)(bn0 + row) * K + k0 + c8);
      *(uint4*)&sBt[cur][row][c8] = val;
    }
#endif
    if (k0 + BK < K && tid < 64)
      __builtin_prefetch(A + (size_t)(bm0 + tid) * K + (k0 + BK), 0, 1);
    __syncthreads();

    v16bf afrag[2];
#pragma unroll
    for (int mt = 0; mt < 2; ++mt) {
      const bf16* ap = &sA[cur][wm * 32 + mt * 16 + lm][0];
      afrag[mt] = load_frag_pair(ap + 8 * half, ap + 16 + 8 * half);
    }
#pragma unroll
    for (int nt = 0; nt < 4; ++nt) {
      const bf16* bp = &sBt[cur][wn * 64 + nt * 16 + lm][0];
      v16bf bfrag = load_frag_pair(bp + 8 * half, bp + 16 + 8 * half);
#pragma unroll
      for (int mt = 0; mt < 2; ++mt)
        acc[mt][nt] = __builtin_amdgcn_wmma_f32_16x16x32_bf16(
            false, afrag[mt], false, bfrag, (short)0, acc[mt][nt], false, false);
    }
    __syncthreads();
  }

#pragma unroll
  for (int mt = 0; mt < 2; ++mt)
#pragma unroll
    for (int nt = 0; nt < 4; ++nt)
#pragma unroll
      for (int r = 0; r < 8; ++r) {
        int gm = bm0 + wm * 32 + mt * 16 + r + 8 * half;
        int gn = bn0 + wn * 64 + nt * 16 + lm;
        if (gn < N) {
          float vv = acc[mt][nt][r];
          if (c_is_bf16) ((bf16*)Cout)[(size_t)gm * N + gn] = (bf16)vv;
          else           ((float*)Cout)[(size_t)gm * N + gn] = vv;
        }
      }
}

// ---------------------------------------------------------------- RoPE + rearrange
// q[H][S][192], k[H][S][192] (key-major), v[H][128][S] (dim-major for P@V B-frags).
__global__ void rope_rearrange(
    const bf16* __restrict__ qC, const bf16* __restrict__ qR,
    const bf16* __restrict__ kC, const bf16* __restrict__ kR,
    const bf16* __restrict__ vin,
    bf16* __restrict__ q, bf16* __restrict__ k, bf16* __restrict__ v)
{
  int s = blockIdx.x;
  for (int idx = threadIdx.x; idx < NH * DQK; idx += blockDim.x) {
    int h = idx / DQK, d = idx % DQK;
    size_t qo = ((size_t)h * S_LEN + s) * DQK + d;
    if (d < HDIM) {
      q[qo] = qC[(size_t)s * (NH * HDIM) + h * HDIM + d];
      k[qo] = kC[(size_t)s * (NH * HDIM) + h * HDIM + d];
      v[((size_t)h * HDIM + d) * S_LEN + s] = vin[(size_t)s * (NH * HDIM) + h * HDIM + d];
    } else {
      int j  = d - HDIM;
      int jj = j & 31;
      float fr = (float)s * __powf(10000.0f, -(float)jj / 32.0f);
      float cs = __cosf(fr), sn = __sinf(fr);
      float x  = (float)qR[(size_t)s * (NH * DRT) + h * DRT + j];
      float xo = (j < 32) ? -(float)qR[(size_t)s * (NH * DRT) + h * DRT + j + 32]
                          :  (float)qR[(size_t)s * (NH * DRT) + h * DRT + j - 32];
      q[qo] = (bf16)(x * cs + xo * sn);
      float y  = (float)kR[(size_t)s * DRT + j];
      float yo = (j < 32) ? -(float)kR[(size_t)s * DRT + j + 32]
                          :  (float)kR[(size_t)s * DRT + j - 32];
      k[qo] = (bf16)(y * cs + yo * sn);
    }
  }
}

// ---------------------------------------------------------------- flash attention (causal)
// Block = 8 waves handles (head, 128 query rows); each wave owns 16 rows.
// K/V tiles DMA'd by the TDM (wave 0), double-buffered so the next slab's DMA
// overlaps the current slab's WMMAs; online softmax; P transposed through
// per-wave LDS scratch for the C-layout -> A-layout change.
__global__ __launch_bounds__(256) void mla_attention(
    const bf16* __restrict__ q, const bf16* __restrict__ k,
    const bf16* __restrict__ v, bf16* __restrict__ attn)
{
  __shared__ __align__(16) bf16 sK[2][32][DQK];     // 2 x 12 KB   [key][dim]
  __shared__ __align__(16) bf16 sVt[2][HDIM][32];   // 2 x 8 KB    [dim][key]
  __shared__ __align__(16) bf16 sP[8][16][32];      // 8 KB per-wave P scratch

  const float scale = 0.07216878364870323f;         // 1/sqrt(192)
  const int h    = blockIdx.y;
  const int q0   = blockIdx.x * 128;
  const int tid  = threadIdx.x;
  const int w    = tid >> 5;
  const int l    = tid & 31;
  const int lm   = l & 15;
  const int half = l >> 4;
  const int qrow_base = q0 + w * 16;

  // Q fragments stay in registers for the whole K sweep (6 x 16x32 bf16).
  v16bf qf[6];
  {
    const bf16* qp = q + ((size_t)h * S_LEN + qrow_base + lm) * DQK;
#pragma unroll
    for (int c = 0; c < 6; ++c)
      qf[c] = load_frag_pair(qp + 32 * c + 8 * half, qp + 32 * c + 16 + 8 * half);
  }

  const v8f zero8 = {0.f,0.f,0.f,0.f,0.f,0.f,0.f,0.f};
  v8f o[8];
#pragma unroll
  for (int j = 0; j < 8; ++j) o[j] = zero8;
  float mrow[8], lrow[8];
#pragma unroll
  for (int r = 0; r < 8; ++r) { mrow[r] = -1e30f; lrow[r] = 0.f; }

  const int niter = (q0 + 128) / 32;                // causal horizon in 32-key slabs

#ifdef HAVE_TDM
  if (w == 0) {   // prologue: DMA slab 0
    tdm_load_2d((unsigned)(size_t)&sK[0][0][0],
                k + (size_t)h * S_LEN * DQK,
                DQK, 32, DQK, (unsigned)S_LEN, (unsigned long)DQK);
    tdm_load_2d((unsigned)(size_t)&sVt[0][0][0],
                v + (size_t)h * HDIM * S_LEN,
                32, HDIM, (unsigned)S_LEN, HDIM, (unsigned long)S_LEN);
  }
#endif

  for (int it = 0; it < niter; ++it) {
    const int cur = it & 1;
    const int kt  = it * 32;
#ifdef HAVE_TDM
    if (w == 0) {
      if (it + 1 < niter) {
        const int nkt = kt + 32;
        tdm_load_2d((unsigned)(size_t)&sK[1 - cur][0][0],
                    k + ((size_t)h * S_LEN + nkt) * DQK,
                    DQK, 32, DQK, (unsigned)(S_LEN - nkt), (unsigned long)DQK);
        tdm_load_2d((unsigned)(size_t)&sVt[1 - cur][0][0],
                    v + (size_t)h * HDIM * S_LEN + nkt,
                    32, HDIM, (unsigned)(S_LEN - nkt), HDIM, (unsigned long)S_LEN);
        __builtin_amdgcn_s_wait_tensorcnt(2);
      } else {
        __builtin_amdgcn_s_wait_tensorcnt(0);
      }
    }
#else
    for (int ch = tid; ch < 32 * (DQK / 8); ch += 256) {
      int row = ch / 24, c8 = (ch % 24) * 8;
      *(uint4*)&sK[cur][row][c8] =
          *(const uint4*)(k + ((size_t)h * S_LEN + kt + row) * DQK + c8);
    }
    for (int ch = tid; ch < (HDIM * 32) / 8; ch += 256) {
      int row = ch >> 2, c8 = (ch & 3) * 8;
      *(uint4*)&sVt[cur][row][c8] =
          *(const uint4*)(v + ((size_t)h * HDIM + row) * S_LEN + kt + c8);
    }
#endif
    __syncthreads();

    // ---- scores: two 16-key subtiles of Q(16x192) x K^T ----
    v8f sc[2];
#pragma unroll
    for (int kk = 0; kk < 2; ++kk) {
      v8f a = zero8;
      const bf16* kp = &sK[cur][kk * 16 + lm][0];   // lane = key column of B
#pragma unroll
      for (int c = 0; c < 6; ++c) {
        v16bf bfrag = load_frag_pair(kp + 32 * c + 8 * half,
                                     kp + 32 * c + 16 + 8 * half);
        a = __builtin_amdgcn_wmma_f32_16x16x32_bf16(
            false, qf[c], false, bfrag, (short)0, a, false, false);
      }
      sc[kk] = a;
    }

    // ---- online softmax over the 32-key slab ----
#pragma unroll
    for (int r = 0; r < 8; ++r) {
      int qrow = qrow_base + r + 8 * half;
      float s0 = sc[0][r] * scale + (((kt      + lm) > qrow) ? -1e9f : 0.f);
      float s1 = sc[1][r] * scale + (((kt + 16 + lm) > qrow) ? -1e9f : 0.f);
      float mx = fmaxf(s0, s1);
#pragma unroll
      for (int off = 8; off; off >>= 1) mx = fmaxf(mx, __shfl_xor(mx, off, 32));
      float mnew = fmaxf(mrow[r], mx);
      float p0 = __expf(s0 - mnew), p1 = __expf(s1 - mnew);
      float rs = p0 + p1;
#pragma unroll
      for (int off = 8; off; off >>= 1) rs += __shfl_xor(rs, off, 32);
      float corr = __expf(mrow[r] - mnew);
      lrow[r] = lrow[r] * corr + rs;
      mrow[r] = mnew;
#pragma unroll
      for (int j = 0; j < 8; ++j) o[j][r] = o[j][r] * corr;
      sP[w][r + 8 * half][lm]      = (bf16)p0;
      sP[w][r + 8 * half][16 + lm] = (bf16)p1;
    }

    // ---- O += P(16x32) x V(32x128) ----
    const bf16* pp = &sP[w][lm][0];
    v16bf pA = load_frag_pair(pp + 8 * half, pp + 16 + 8 * half);
#pragma unroll
    for (int j = 0; j < 8; ++j) {
      const bf16* vp = &sVt[cur][16 * j + lm][0];   // lane = dim column of B
      v16bf vB = load_frag_pair(vp + 8 * half, vp + 16 + 8 * half);
      o[j] = __builtin_amdgcn_wmma_f32_16x16x32_bf16(
          false, pA, false, vB, (short)0, o[j], false, false);
    }
    __syncthreads();
  }

  // ---- normalize + store attn[S][H*128] bf16 (A-operand of W_O GEMM) ----
#pragma unroll
  for (int j = 0; j < 8; ++j)
#pragma unroll
    for (int r = 0; r < 8; ++r) {
      int qrow = qrow_base + r + 8 * half;
      attn[(size_t)qrow * (NH * HDIM) + h * HDIM + 16 * j + lm] =
          (bf16)(o[j][r] / lrow[r]);
    }
}

// ---------------------------------------------------------------- host orchestration
extern "C" void kernel_launch(void* const* d_in, const int* in_sizes, int n_in,
                              void* d_out, int out_size, void* d_ws, size_t ws_size,
                              hipStream_t stream) {
  (void)in_sizes; (void)n_in; (void)out_size; (void)ws_size;
  const float* hidden = (const float*)d_in[0];
  /* d_in[1] = mask: causal mask is analytic, never read */
  const float* W_DKV = (const float*)d_in[2];
  const float* W_UK  = (const float*)d_in[3];
  const float* W_UV  = (const float*)d_in[4];
  const float* W_DQ  = (const float*)d_in[5];
  const float* W_UQ  = (const float*)d_in[6];
  const float* W_QR  = (const float*)d_in[7];
  const float* W_KR  = (const float*)d_in[8];
  const float* W_O   = (const float*)d_in[9];
  float* out = (float*)d_out;

  char* p = (char*)d_ws;
  auto alloc = [&](size_t elems) -> bf16* {
    bf16* r = (bf16*)p;
    p += (elems * sizeof(bf16) + 255) & ~(size_t)255;
    return r;
  };
  bf16* hbf  = alloc((size_t)S_LEN * HID);
  bf16* wdkv = alloc((size_t)HID * CKV_D);          // transposed [N][K]
  bf16* wuk  = alloc((size_t)CKV_D * (NH * HDIM));
  bf16* wuv  = alloc((size_t)CKV_D * (NH * HDIM));
  bf16* wdq  = alloc((size_t)HID * CQ_D);
  bf16* wuq  = alloc((size_t)CQ_D * (NH * HDIM));
  bf16* wqr  = alloc((size_t)CQ_D * (NH * DRT));
  bf16* wkr  = alloc((size_t)HID * DRT);
  bf16* wo   = alloc((size_t)HID * HID);
  bf16* ckv  = alloc((size_t)S_LEN * CKV_D);
  bf16* cq   = alloc((size_t)S_LEN * CQ_D);
  bf16* kc   = alloc((size_t)S_LEN * (NH * HDIM));
  bf16* vt   = alloc((size_t)S_LEN * (NH * HDIM));
  bf16* qc   = alloc((size_t)S_LEN * (NH * HDIM));
  bf16* qr   = alloc((size_t)S_LEN * (NH * DRT));
  bf16* kr   = alloc((size_t)S_LEN * DRT);
  bf16* qbuf = alloc((size_t)NH * S_LEN * DQK);
  bf16* kbuf = alloc((size_t)NH * S_LEN * DQK);
  bf16* vbuf = alloc((size_t)NH * HDIM * S_LEN);
  bf16* attb = alloc((size_t)S_LEN * HID);

  cast_f32_bf16<<<(S_LEN * HID + 255) / 256, 256, 0, stream>>>(hidden, hbf, S_LEN * HID);
  auto castT = [&](const float* s, bf16* d, int R, int C) {   // [R][C] -> [C][R]
    cast_transpose_f32_bf16<<<(R * C + 255) / 256, 256, 0, stream>>>(s, d, R, C);
  };
  castT(W_DKV, wdkv, HID,   CKV_D);
  castT(W_UK,  wuk,  CKV_D, NH * HDIM);
  castT(W_UV,  wuv,  CKV_D, NH * HDIM);
  castT(W_DQ,  wdq,  HID,   CQ_D);
  castT(W_UQ,  wuq,  CQ_D,  NH * HDIM);
  castT(W_QR,  wqr,  CQ_D,  NH * DRT);
  castT(W_KR,  wkr,  HID,   DRT);
  castT(W_O,   wo,   HID,   HID);

  auto gemm = [&](const bf16* A, const bf16* Bt, void* C, int M, int N, int K, int obf) {
    dim3 g((N + BN - 1) / BN, M / BM);
    gemm_bf16_kernel<<<g, 256, 0, stream>>>(A, Bt, C, M, N, K, obf);
  };
  gemm(hbf, wdkv, ckv, S_LEN, CKV_D,     HID,   1);   // c_KV
  gemm(ckv, wuk,  kc,  S_LEN, NH * HDIM, CKV_D, 1);   // k_C
  gemm(ckv, wuv,  vt,  S_LEN, NH * HDIM, CKV_D, 1);   // v
  gemm(hbf, wdq,  cq,  S_LEN, CQ_D,      HID,   1);   // c_Q
  gemm(cq,  wuq,  qc,  S_LEN, NH * HDIM, CQ_D,  1);   // q_C
  gemm(cq,  wqr,  qr,  S_LEN, NH * DRT,  CQ_D,  1);   // q_R
  gemm(hbf, wkr,  kr,  S_LEN, DRT,       HID,   1);   // k_R

  rope_rearrange<<<S_LEN, 256, 0, stream>>>(qc, qr, kc, kr, vt, qbuf, kbuf, vbuf);

  mla_attention<<<dim3(S_LEN / 128, NH), 256, 0, stream>>>(qbuf, kbuf, vbuf, attb);

  gemm(attb, wo, out, S_LEN, HID, HID, 0);            // final projection (fp32)
}
